// Famnet_42253888258423
// MI455X (gfx1250) — matrix operand order repaired and checked by reading, not compiled, not run on hardware
//
#include <hip/hip_runtime.h>
#include <hip/hip_bf16.h>

typedef _Float16 v8h  __attribute__((ext_vector_type(8)));
typedef _Float16 v16h __attribute__((ext_vector_type(16)));
typedef float    v8f  __attribute__((ext_vector_type(8)));

#define DEV static __device__ __forceinline__

DEV int imin(int a, int b) { return a < b ? a : b; }
DEV int imax(int a, int b) { return a > b ? a : b; }

static constexpr int MAXP = 24;   // upper bound on patch side (boxes <= 96 px, /8 + slack)

struct Geom {
    int top[3], left[3], bot[3], right[3];
    int PH, PW;
    int PHs[3], PWs[3];
};

// ---------------- workspace layout (bytes) ----------------
static constexpr size_t AL(size_t x) { return (x + 255) & ~(size_t)255; }
static constexpr size_t OFF_GEOM  = 0;
static constexpr size_t OFF_ACT1  = AL(OFF_GEOM + 1024);                         // 2x192x192x64   f16 NHWC
static constexpr size_t OFF_ACT2  = AL(OFF_ACT1 + (size_t)2*192*192*64*2);       // 2x96x96x256    f16 NHWC
static constexpr size_t OFF_F3H   = AL(OFF_ACT2 + (size_t)2*96*96*256*2);        // 2x48x48x512    f16 NHWC
static constexpr size_t OFF_F3F   = AL(OFF_F3H  + (size_t)2*48*48*512*2);        // 2x512x48x48    f32 NCHW
static constexpr size_t OFF_F4F   = AL(OFF_F3F  + (size_t)2*512*48*48*4);        // 2x1024x24x24   f32 NCHW
static constexpr size_t OFF_WP2   = AL(OFF_F4F  + (size_t)2*1024*24*24*4);       // 9x256x64  f16
static constexpr size_t OFF_WP3   = AL(OFF_WP2  + (size_t)9*256*64*2);           // 9x512x256 f16
static constexpr size_t OFF_WP4   = AL(OFF_WP3  + (size_t)9*512*256*2);          // 9x1024x512 f16
static constexpr size_t OFF_WR1   = AL(OFF_WP4  + (size_t)9*1024*512*2);         // 49x224x32 f16
static constexpr size_t OFF_WR2   = AL(OFF_WR1  + (size_t)49*224*32*2);          // 25x128x224 f16
static constexpr size_t OFF_WR3   = AL(OFF_WR2  + (size_t)25*128*224*2);         // 9x64x128 f16
static constexpr size_t OFF_PATCH = AL(OFF_WR3  + (size_t)9*64*128*2);           // 3x1024xMAXPxMAXP f32
static constexpr size_t OFF_PF    = AL(OFF_PATCH+ (size_t)3*1024*MAXP*MAXP*4);   // scaled patches
static constexpr size_t OFF_SIM   = AL(OFF_PF   + (size_t)3*1024*MAXP*MAXP*4);   // 3x48x48 f32
static constexpr size_t OFF_X     = AL(OFF_SIM  + (size_t)3*48*48*4);            // 6x48x48x32 f16 NHWC (6ch + pad)
static constexpr size_t OFF_R1    = AL(OFF_X    + (size_t)6*48*48*32*2);         // 6x48x48x224
static constexpr size_t OFF_U1    = AL(OFF_R1   + (size_t)6*48*48*224*2);        // 6x96x96x224
static constexpr size_t OFF_R2    = AL(OFF_U1   + (size_t)6*96*96*224*2);        // 6x96x96x128
static constexpr size_t OFF_U2    = AL(OFF_R2   + (size_t)6*96*96*128*2);        // 6x192x192x128
static constexpr size_t OFF_R3    = AL(OFF_U2   + (size_t)6*192*192*128*2);      // 6x192x192x64
static constexpr size_t WS_NEEDED = OFF_R3 + (size_t)6*192*192*64*2;

// ---------------- half-pixel bilinear coordinate ----------------
DEV void hp_coord(int i, int n_out, int n_in, int& i0, int& i1, float& t)
{
    float src = ((float)i + 0.5f) * ((float)n_in / (float)n_out) - 0.5f;
    src = fminf(fmaxf(src, 0.0f), (float)(n_in - 1));
    i0 = (int)floorf(src);
    if (i0 > n_in - 1) i0 = n_in - 1;
    t  = src - (float)i0;
    i1 = imin(i0 + 1, n_in - 1);
}

// ---------------- weight packing: OIHW f32 -> [kh][kw][coPad][ciPad] f16 ----------------
__global__ void pack_w_kernel(const float* __restrict__ w, _Float16* __restrict__ wp,
                              int Co, int Ci, int KH, int KW, int CoPad, int CiPad)
{
    int i = blockIdx.x * blockDim.x + threadIdx.x;
    int total = KH * KW * CoPad * CiPad;
    if (i >= total) return;
    int ci = i % CiPad; int r = i / CiPad;
    int co = r % CoPad; int k = r / CoPad;
    int kh = k / KW, kw = k % KW;
    float v = 0.0f;
    if (co < Co && ci < Ci)
        v = w[(((size_t)co * Ci + ci) * KH + kh) * KW + kw];
    wp[i] = (_Float16)v;
}

// ---------------- conv1: 3->64 7x7 s2 p3 direct f32, NCHW f32 -> NHWC f16, relu ----------------
__global__ void conv1_kernel(const float* __restrict__ img, const float* __restrict__ w,
                             _Float16* __restrict__ out)
{
    int i = blockIdx.x * blockDim.x + threadIdx.x;
    if (i >= 2 * 192 * 192 * 64) return;
    int co = i & 63; int r = i >> 6;
    int ow = r % 192; r /= 192;
    int oh = r % 192; int b = r / 192;
    float s = 0.0f;
    for (int ci = 0; ci < 3; ++ci) {
        const float* ip = img + (size_t)(b * 3 + ci) * 384 * 384;
        const float* wk = w + (size_t)(co * 3 + ci) * 49;
        #pragma unroll
        for (int kh = 0; kh < 7; ++kh) {
            int ih = oh * 2 - 3 + kh;
            if ((unsigned)ih >= 384u) continue;
            #pragma unroll
            for (int kw = 0; kw < 7; ++kw) {
                int iw = ow * 2 - 3 + kw;
                if ((unsigned)iw < 384u) s += ip[ih * 384 + iw] * wk[kh * 7 + kw];
            }
        }
    }
    out[((size_t)(b * 192 + oh) * 192 + ow) * 64 + co] = (_Float16)fmaxf(s, 0.0f);
}

// ---------------- implicit-GEMM conv on WMMA f32_16x16x32_f16, LDS-staged B ----------------
// Block = 128 threads = 4 waves. One 16-pixel tile per block; wave w handles co-tile
// blockIdx.y*4+w. The 16 im2col rows (16 x CinPad f16) are staged in LDS once per
// (kh,kw) tap by all 128 threads and consumed by all 4 waves (4x global B reuse).
// in:  [img][Hin][Win][CinPad] f16 (CinPad % 32 == 0, CinPad <= 512)
// wp:  [KH*KW][CoPad][CinPad]  f16
// out16: [img][Hout][Wout][CoPad] f16 (relu)   out32: [img][CoReal][Hout][Wout] f32 (relu)
template <int KH, int KW, int STRIDE, int PAD, bool W16, bool W32>
__global__ void __launch_bounds__(128)
conv_wmma_kernel(const _Float16* __restrict__ in, const _Float16* __restrict__ wp,
                 const float* __restrict__ bias,
                 _Float16* __restrict__ out16, float* __restrict__ out32,
                 int Hin, int Win, int CinPad, int Hout, int Wout, int CoPad, int CoReal)
{
    __shared__ _Float16 smem[16 * 512];   // 16 KB of the 320 KB WGP LDS

    const int t    = threadIdx.x;         // 0..127
    const int lane = t & 31;
    const int wv   = t >> 5;              // wave id 0..3
    const int m    = lane & 15;           // A row (co) == B col (pixel) for this lane
    const int hh   = lane >> 4;           // lane half selects K sub-range
    const int co0  = (blockIdx.y * 4 + wv) * 16;
    const bool wvAct = co0 < CoPad;       // wave-uniform (EXEC stays all-ones for WMMA)
    const int img  = blockIdx.z;
    const int npix = Hout * Wout;
    const int pix0 = blockIdx.x * 16;

    // compute-phase pixel for this lane
    const int pix = pix0 + m;
    const bool vpix = pix < npix;
    const int oh = vpix ? (pix / Wout) : 0;
    const int ow = vpix ? (pix % Wout) : 0;

    // staging-phase row for this thread: row rowN (0..15), 8 threads per row
    const int rowN  = t >> 3;
    const int lane8 = t & 7;
    const int pixS  = pix0 + rowN;
    const bool vS   = pixS < npix;
    const int ohS = vS ? (pixS / Wout) : 0;
    const int owS = vS ? (pixS % Wout) : 0;

    const _Float16* inImg = in + (size_t)img * Hin * Win * CinPad;

    v8f acc = {};
    for (int kh = 0; kh < KH; ++kh) {
        const int ihS = ohS * STRIDE - PAD + kh;
        const bool vyS = vS && ((unsigned)ihS < (unsigned)Hin);
        for (int kw = 0; kw < KW; ++kw) {
            // ---- stage B rows into LDS (all 128 threads) ----
            const int iwS = owS * STRIDE - PAD + kw;
            const bool vSrc = vyS && ((unsigned)iwS < (unsigned)Win);
            const _Float16* srow = inImg + ((size_t)ihS * Win + iwS) * CinPad;
            _Float16* drow = smem + rowN * CinPad;
            for (int off = lane8 * 8; off < CinPad; off += 64) {
                v8h val = {};
                if (vSrc) {
                    val = *(const v8h*)(srow + off);
                    __builtin_prefetch(srow + off + 64, 0, 0);
                }
                *(v8h*)(drow + off) = val;
            }
            __syncthreads();

            // ---- consume: each wave does its co-tile ----
            if (wvAct) {
                const _Float16* aRow =
                    wp + ((size_t)(kh * KW + kw) * CoPad + (co0 + m)) * CinPad;
                const _Float16* bRow = smem + m * CinPad;    // pixel n == m for this lane
                for (int ck = 0; ck < CinPad; ck += 32) {
                    // A (16x32 f16): lane holds contiguous K run [ck+16h..+15]
                    v16h A = *(const v16h*)(aRow + ck + 16 * hh);
                    // B (32x16 f16): lane holds contiguous K run [ck+16h..+15] from LDS
                    v16h B = *(const v16h*)(bRow + ck + 16 * hh);
                    acc = __builtin_amdgcn_wmma_f32_16x16x32_f16(
                              false, A, false, B, (short)0, acc, false, false);
                }
            }
            __syncthreads();
        }
    }

    if (wvAct && vpix) {
        v8h o = {};
        #pragma unroll
        for (int r = 0; r < 8; ++r) {
            const int co = co0 + r + 8 * hh;  // C/D layout: VGPR r -> M = r + 8*(lane>=16)
            float val = acc[r];
            if (bias && co < CoReal) val += bias[co];
            val = fmaxf(val, 0.0f);
            o[r] = (_Float16)val;
            if constexpr (W32) {
                if (co < CoReal)
                    out32[(((size_t)img * CoReal + co) * Hout + oh) * Wout + ow] = val;
            }
        }
        if constexpr (W16) {
            *(v8h*)(out16 + ((size_t)img * npix + pix) * CoPad + co0 + 8 * hh) = o;
        }
    }
}

// ---------------- box geometry from tlbrs (device side, 4 threads: (b, level)) ----------------
__global__ void geom_kernel(const float* __restrict__ tlbrs, Geom* __restrict__ g)
{
    int t = threadIdx.x;
    if (t >= 4) return;
    int b = t >> 1, level = t & 1;
    const int FH = level ? 24 : 48;
    const float scaling = 384.0f / (float)FH;
    Geom G;
    int PH = 1, PW = 1;
    for (int p = 0; p < 3; ++p) {
        const float* bx = tlbrs + ((size_t)b * 3 + p) * 4;
        int top  = imax((int)floorf(bx[0] / scaling), 0);
        int left = imax((int)floorf(bx[1] / scaling), 0);
        int bot  = imin((int)ceilf(bx[2] / scaling) + 1, FH);
        int rgt  = imin((int)ceilf(bx[3] / scaling) + 1, FH);
        G.top[p] = top; G.left[p] = left; G.bot[p] = bot; G.right[p] = rgt;
        PH = imax(PH, bot - top);
        PW = imax(PW, rgt - left);
    }
    PH = imin(PH, MAXP); PW = imin(PW, MAXP);
    G.PH = PH; G.PW = PW;
    const float scales[3] = {1.0f, 0.9f, 1.1f};
    for (int s = 0; s < 3; ++s) {
        int PHs = (int)ceilf((float)PH * scales[s]); if (PHs < 1) PHs = PH;
        int PWs = (int)ceilf((float)PW * scales[s]); if (PWs < 1) PWs = PW;
        G.PHs[s] = imin(PHs, MAXP);
        G.PWs[s] = imin(PWs, MAXP);
    }
    g[t] = G;
}

// ---------------- patch extraction: crop + half-pixel resize to (PH, PW) ----------------
__global__ void patch_extract_kernel(const float* __restrict__ fm, const Geom* __restrict__ g,
                                     float* __restrict__ patches, int FC, int FH, int FW)
{
    int i = blockIdx.x * blockDim.x + threadIdx.x;
    int x = i % MAXP; int r = i / MAXP;
    int y = r % MAXP; r /= MAXP;
    int c = r % FC;   int p = r / FC;
    if (p >= 3) return;
    int PH = g->PH, PW = g->PW;
    if (y >= PH || x >= PW) return;
    int top = g->top[p], left = g->left[p];
    int in_h = g->bot[p] - top, in_w = g->right[p] - left;
    int y0, y1, x0, x1; float ty, tx;
    hp_coord(y, PH, in_h, y0, y1, ty);
    hp_coord(x, PW, in_w, x0, x1, tx);
    const float* f = fm + (size_t)c * FH * FW;
    float v00 = f[(top + y0) * FW + left + x0];
    float v01 = f[(top + y0) * FW + left + x1];
    float v10 = f[(top + y1) * FW + left + x0];
    float v11 = f[(top + y1) * FW + left + x1];
    float v = (1.f - ty) * ((1.f - tx) * v00 + tx * v01)
            +        ty  * ((1.f - tx) * v10 + tx * v11);
    patches[((size_t)(p * FC + c) * PH + y) * PW + x] = v;
}

// ---------------- patch rescale to (PHs, PWs) for one scale ----------------
__global__ void patch_scale_kernel(const float* __restrict__ patches, const Geom* __restrict__ g,
                                   float* __restrict__ pf, int FC, int si)
{
    int i = blockIdx.x * blockDim.x + threadIdx.x;
    int x = i % MAXP; int r = i / MAXP;
    int y = r % MAXP; r /= MAXP;
    int c = r % FC;   int p = r / FC;
    if (p >= 3) return;
    int PH = g->PH, PW = g->PW, PHs = g->PHs[si], PWs = g->PWs[si];
    if (y >= PHs || x >= PWs) return;
    int y0, y1, x0, x1; float ty, tx;
    hp_coord(y, PHs, PH, y0, y1, ty);
    hp_coord(x, PWs, PW, x0, x1, tx);
    const float* s = patches + (size_t)(p * FC + c) * PH * PW;
    float v = (1.f - ty) * ((1.f - tx) * s[y0 * PW + x0] + tx * s[y0 * PW + x1])
            +        ty  * ((1.f - tx) * s[y1 * PW + x0] + tx * s[y1 * PW + x1]);
    pf[((size_t)(p * FC + c) * PHs + y) * PWs + x] = v;
}

// ---------------- correlation: sim[p] = fm (*) pf[p], pad (PHs/2, (PHs-1)/2) ----------------
__global__ void corr_kernel(const float* __restrict__ fm, const float* __restrict__ pf,
                            const Geom* __restrict__ g, float* __restrict__ sim,
                            int FC, int FH, int FW, int si)
{
    int i = blockIdx.x * blockDim.x + threadIdx.x;
    int ow = i % FW; int r = i / FW;
    int oh = r % FH; int p = r / FH;
    if (p >= 3) return;
    int PHs = g->PHs[si], PWs = g->PWs[si];
    int padT = PHs >> 1, padL = PWs >> 1;
    float s = 0.0f;
    for (int c = 0; c < FC; ++c) {
        const float* fmc = fm + (size_t)c * FH * FW;
        const float* pfc = pf + (size_t)(p * FC + c) * PHs * PWs;
        for (int kh = 0; kh < PHs; ++kh) {
            int ih = oh - padT + kh;
            if ((unsigned)ih >= (unsigned)FH) continue;
            const float* frow = fmc + ih * FW;
            const float* prow = pfc + kh * PWs;
            for (int kw = 0; kw < PWs; ++kw) {
                int iw = ow - padL + kw;
                if ((unsigned)iw < (unsigned)FW) s += frow[iw] * prow[kw];
            }
        }
    }
    sim[((size_t)p * FH + oh) * FW + ow] = s;
}

// ---------------- sim (FH,FW) -> regressor input x (48,48) NHWC f16, channel level*3+si ----------------
__global__ void sim_to_x_kernel(const float* __restrict__ sim, _Float16* __restrict__ xbuf,
                                int FH, int FW, int b, int level, int si)
{
    int i = blockIdx.x * blockDim.x + threadIdx.x;
    int x = i % 48; int r = i / 48;
    int y = r % 48; int p = r / 48;
    if (p >= 3) return;
    float v;
    if (FH == 48) {
        v = sim[((size_t)p * 48 + y) * 48 + x];
    } else {
        int y0, y1, x0, x1; float ty, tx;
        hp_coord(y, 48, FH, y0, y1, ty);
        hp_coord(x, 48, FW, x0, x1, tx);
        const float* s = sim + (size_t)p * FH * FW;
        v = (1.f - ty) * ((1.f - tx) * s[y0 * FW + x0] + tx * s[y0 * FW + x1])
          +        ty  * ((1.f - tx) * s[y1 * FW + x0] + tx * s[y1 * FW + x1]);
    }
    int img = b * 3 + p;
    int ch = level * 3 + si;
    xbuf[((size_t)(img * 48 + y) * 48 + x) * 32 + ch] = (_Float16)v;
}

// ---------------- align-corners 2x bilinear upsample, NHWC f16 ----------------
__global__ void upsample_ac_kernel(const _Float16* __restrict__ in, _Float16* __restrict__ out,
                                   int imgs, int Hin, int C)
{
    int Hout = Hin * 2;
    int i = blockIdx.x * blockDim.x + threadIdx.x;
    int c = i % C; int r = i / C;
    int x = r % Hout; r /= Hout;
    int y = r % Hout; int img = r / Hout;
    if (img >= imgs) return;
    float py = (float)y * (float)(Hin - 1) / (float)(Hout - 1);
    int y0 = (int)py; float ty = py - (float)y0; int y1 = imin(y0 + 1, Hin - 1);
    float px = (float)x * (float)(Hin - 1) / (float)(Hout - 1);
    int x0 = (int)px; float tx = px - (float)x0; int x1 = imin(x0 + 1, Hin - 1);
    const _Float16* base = in + (size_t)img * Hin * Hin * C;
    float v00 = (float)base[((size_t)y0 * Hin + x0) * C + c];
    float v01 = (float)base[((size_t)y0 * Hin + x1) * C + c];
    float v10 = (float)base[((size_t)y1 * Hin + x0) * C + c];
    float v11 = (float)base[((size_t)y1 * Hin + x1) * C + c];
    float v = (1.f - ty) * ((1.f - tx) * v00 + tx * v01)
            +        ty  * ((1.f - tx) * v10 + tx * v11);
    out[(((size_t)img * Hout + y) * Hout + x) * C + c] = (_Float16)v;
}

// ---------------- fused: upsample2x_ac(192->384) + 1x1 conv 64->32 relu + 1x1 conv 32->1 relu + max over P ----
__global__ void final_kernel(const _Float16* __restrict__ r3,
                             const float* __restrict__ rw4, const float* __restrict__ rb4,
                             const float* __restrict__ rw5, const float* __restrict__ rb5,
                             float* __restrict__ out)
{
    int i = blockIdx.x * blockDim.x + threadIdx.x;
    if (i >= 2 * 384 * 384) return;
    int x = i % 384; int r = i / 384;
    int y = r % 384; int b = r / 384;
    float py = (float)y * 191.0f / 383.0f;
    int y0 = (int)py; float ty = py - (float)y0; int y1 = imin(y0 + 1, 191);
    float px = (float)x * 191.0f / 383.0f;
    int x0 = (int)px; float tx = px - (float)x0; int x1 = imin(x0 + 1, 191);
    float w00 = (1.f - ty) * (1.f - tx), w01 = (1.f - ty) * tx;
    float w10 = ty * (1.f - tx),        w11 = ty * tx;
    float best = 0.0f;   // outputs are relu'd (>= 0)
    for (int p = 0; p < 3; ++p) {
        const _Float16* base = r3 + (size_t)(b * 3 + p) * 192 * 192 * 64;
        const _Float16* p00 = base + ((size_t)y0 * 192 + x0) * 64;
        const _Float16* p01 = base + ((size_t)y0 * 192 + x1) * 64;
        const _Float16* p10 = base + ((size_t)y1 * 192 + x0) * 64;
        const _Float16* p11 = base + ((size_t)y1 * 192 + x1) * 64;
        float h[64];
        #pragma unroll
        for (int c = 0; c < 64; ++c)
            h[c] = w00 * (float)p00[c] + w01 * (float)p01[c]
                 + w10 * (float)p10[c] + w11 * (float)p11[c];
        float acc5 = rb5[0];
        for (int j = 0; j < 32; ++j) {
            float a = rb4[j];
            const float* wr = rw4 + j * 64;
            #pragma unroll
            for (int c = 0; c < 64; ++c) a += wr[c] * h[c];
            acc5 += rw5[j] * fmaxf(a, 0.0f);
        }
        best = fmaxf(best, fmaxf(acc5, 0.0f));
    }
    out[i] = best;
}

__global__ void zero_f16_kernel(_Float16* __restrict__ p, int n)
{
    int i = blockIdx.x * blockDim.x + threadIdx.x;
    if (i < n) p[i] = (_Float16)0.0f;
}

static inline dim3 G(long n, int bs) { return dim3((unsigned)((n + bs - 1) / bs)); }

extern "C" void kernel_launch(void* const* d_in, const int* in_sizes, int n_in,
                              void* d_out, int out_size, void* d_ws, size_t ws_size,
                              hipStream_t stream)
{
    (void)in_sizes; (void)n_in; (void)out_size;
    if (ws_size < WS_NEEDED) return;   // ~198 MB of scratch required

    const float* images = (const float*)d_in[0];
    const float* tlbrs  = (const float*)d_in[1];
    const float* fw1 = (const float*)d_in[2];
    const float* fw2 = (const float*)d_in[3];
    const float* fw3 = (const float*)d_in[4];
    const float* fw4 = (const float*)d_in[5];
    const float* rw1 = (const float*)d_in[6];
    const float* rb1 = (const float*)d_in[7];
    const float* rw2 = (const float*)d_in[8];
    const float* rb2 = (const float*)d_in[9];
    const float* rw3 = (const float*)d_in[10];
    const float* rb3 = (const float*)d_in[11];
    const float* rw4 = (const float*)d_in[12];
    const float* rb4 = (const float*)d_in[13];
    const float* rw5 = (const float*)d_in[14];
    const float* rb5 = (const float*)d_in[15];
    float* out = (float*)d_out;
    char* ws = (char*)d_ws;

    Geom*      GEOM = (Geom*)     (ws + OFF_GEOM);
    _Float16*  ACT1 = (_Float16*) (ws + OFF_ACT1);
    _Float16*  ACT2 = (_Float16*) (ws + OFF_ACT2);
    _Float16*  F3H  = (_Float16*) (ws + OFF_F3H);
    float*     F3F  = (float*)    (ws + OFF_F3F);
    float*     F4F  = (float*)    (ws + OFF_F4F);
    _Float16*  WP2  = (_Float16*) (ws + OFF_WP2);
    _Float16*  WP3  = (_Float16*) (ws + OFF_WP3);
    _Float16*  WP4  = (_Float16*) (ws + OFF_WP4);
    _Float16*  WR1  = (_Float16*) (ws + OFF_WR1);
    _Float16*  WR2  = (_Float16*) (ws + OFF_WR2);
    _Float16*  WR3  = (_Float16*) (ws + OFF_WR3);
    float*     PATCH= (float*)    (ws + OFF_PATCH);
    float*     PF   = (float*)    (ws + OFF_PF);
    float*     SIM  = (float*)    (ws + OFF_SIM);
    _Float16*  XB   = (_Float16*) (ws + OFF_X);
    _Float16*  R1O  = (_Float16*) (ws + OFF_R1);
    _Float16*  U1   = (_Float16*) (ws + OFF_U1);
    _Float16*  R2O  = (_Float16*) (ws + OFF_R2);
    _Float16*  U2   = (_Float16*) (ws + OFF_U2);
    _Float16*  R3O  = (_Float16*) (ws + OFF_R3);

    // ---- weight packing (f32 OIHW -> f16 [k][coPad][ciPad]) ----
    pack_w_kernel<<<G(9L*256*64, 256), 256, 0, stream>>>(fw2, WP2, 256, 64, 3, 3, 256, 64);
    pack_w_kernel<<<G(9L*512*256, 256), 256, 0, stream>>>(fw3, WP3, 512, 256, 3, 3, 512, 256);
    pack_w_kernel<<<G(9L*1024*512, 256), 256, 0, stream>>>(fw4, WP4, 1024, 512, 3, 3, 1024, 512);
    pack_w_kernel<<<G(49L*224*32, 256), 256, 0, stream>>>(rw1, WR1, 196, 6, 7, 7, 224, 32);
    pack_w_kernel<<<G(25L*128*224, 256), 256, 0, stream>>>(rw2, WR2, 128, 196, 5, 5, 128, 224);
    pack_w_kernel<<<G(9L*64*128, 256), 256, 0, stream>>>(rw3, WR3, 64, 128, 3, 3, 64, 128);

    // ---- backbone ----
    conv1_kernel<<<G(2L*192*192*64, 256), 256, 0, stream>>>(images, fw1, ACT1);
    conv_wmma_kernel<3,3,2,1,true,false><<<dim3(576, 4, 2), 128, 0, stream>>>(
        ACT1, WP2, nullptr, ACT2, nullptr, 192, 192, 64, 96, 96, 256, 256);
    conv_wmma_kernel<3,3,2,1,true,true><<<dim3(144, 8, 2), 128, 0, stream>>>(
        ACT2, WP3, nullptr, F3H, F3F, 96, 96, 256, 48, 48, 512, 512);
    conv_wmma_kernel<3,3,2,1,false,true><<<dim3(36, 16, 2), 128, 0, stream>>>(
        F3H, WP4, nullptr, nullptr, F4F, 48, 48, 512, 24, 24, 1024, 1024);

    // ---- similarity features ----
    geom_kernel<<<1, 32, 0, stream>>>(tlbrs, GEOM);
    zero_f16_kernel<<<G(6L*48*48*32, 256), 256, 0, stream>>>(XB, 6 * 48 * 48 * 32);

    for (int b = 0; b < 2; ++b) {
        for (int level = 0; level < 2; ++level) {
            const int FH = level ? 24 : 48;
            const int FC = level ? 1024 : 512;
            const float* fm = level ? (F4F + (size_t)b * 1024 * 24 * 24)
                                    : (F3F + (size_t)b * 512 * 48 * 48);
            const Geom* g = GEOM + (b * 2 + level);
            long n = 3L * FC * MAXP * MAXP;
            patch_extract_kernel<<<G(n, 256), 256, 0, stream>>>(fm, g, PATCH, FC, FH, FH);
            for (int si = 0; si < 3; ++si) {
                patch_scale_kernel<<<G(n, 256), 256, 0, stream>>>(PATCH, g, PF, FC, si);
                corr_kernel<<<G(3L * FH * FH, 128), 128, 0, stream>>>(fm, PF, g, SIM, FC, FH, FH, si);
                sim_to_x_kernel<<<G(3L * 48 * 48, 256), 256, 0, stream>>>(SIM, XB, FH, FH, b, level, si);
            }
        }
    }

    // ---- count regressor ----
    conv_wmma_kernel<7,7,1,3,true,false><<<dim3(144, 4, 6), 128, 0, stream>>>(
        XB, WR1, rb1, R1O, nullptr, 48, 48, 32, 48, 48, 224, 196);
    upsample_ac_kernel<<<G(6L*96*96*224, 256), 256, 0, stream>>>(R1O, U1, 6, 48, 224);
    conv_wmma_kernel<5,5,1,2,true,false><<<dim3(576, 2, 6), 128, 0, stream>>>(
        U1, WR2, rb2, R2O, nullptr, 96, 96, 224, 96, 96, 128, 128);
    upsample_ac_kernel<<<G(6L*192*192*128, 256), 256, 0, stream>>>(R2O, U2, 6, 96, 128);
    conv_wmma_kernel<3,3,1,1,true,false><<<dim3(2304, 1, 6), 128, 0, stream>>>(
        U2, WR3, rb3, R3O, nullptr, 192, 192, 128, 192, 192, 64, 64);
    final_kernel<<<G(2L*384*384, 128), 128, 0, stream>>>(R3O, rw4, rb4, rw5, rb5, out);
}
